// MinRNN_77292231459589
// MI455X (gfx1250) — matrix operand order repaired and checked by reading, not compiled, not run on hardware
//
#include <hip/hip_runtime.h>
#include <hip/hip_bf16.h>

// ---------------------------------------------------------------------------
// MinRNN on MI455X (gfx1250, wave32, WMMA) — scan fused into the GEMM kernel.
//
// Sizes: B=64, T=2048, E=128, U=128, V=50257.
//   K0a: emb f32 -> f16 (one-time; table then lives in 192MB L2)
//   K0b: W_f/W_i/W_h f32 -> f16, transposed to [gate][U][E]
//   K1 : per 16-timestep tile: gather -> 3x GEMM (v_wmma_f32_16x16x32_f16)
//        -> sigmoid/gate epilogue -> IN-REGISTER 16-step affine composition
//        (h_t = f' h_{t-1} + inj  ==>  (F,I) compose) -> 1KB/block output.
//        Kills the 256MB f'/inj intermediate (now 8MB of chunk affines).
//   K2 : combine 8 consecutive chunk affines (131072 threads)
//   K3 : combine 16 group affines -> h_last = I_total (h0 = 0)
//   K4 : classifier -> sigmoid -> out[B]
// ---------------------------------------------------------------------------

typedef _Float16 v16h __attribute__((ext_vector_type(16)));
typedef _Float16 v8h  __attribute__((ext_vector_type(8)));
typedef float    v8f  __attribute__((ext_vector_type(8)));

#define Bc 64
#define Tc 2048
#define Ec 128
#define Uc 128
#define Vc 50257
#define NCH   128     // Tc/16 chunk affines per batch row
#define NGRP  16      // NCH/8 groups after first combine

union AFrag { v16h v; v8h h[2]; };

// --- K0a: emb f32 -> f16 ----------------------------------------------------
__global__ __launch_bounds__(256) void cvt_emb_kernel(const float* __restrict__ src,
                                                      _Float16* __restrict__ dst, int n) {
    int i = blockIdx.x * 256 + threadIdx.x;
    if (i < n) dst[i] = (_Float16)src[i];
}

// --- K0b: weights f32 [E,U] -> f16 transposed [gate][U][E] ------------------
__global__ __launch_bounds__(256) void cvt_w_kernel(const float* __restrict__ Wf,
                                                    const float* __restrict__ Wi,
                                                    const float* __restrict__ Wh,
                                                    _Float16* __restrict__ wt) {
    int i = blockIdx.x * 256 + threadIdx.x;      // 3*U*E = 49152 threads
    int g = i >> 14;
    int r = i & 16383;
    int n = r >> 7;                               // U index
    int k = r & 127;                              // E index
    const float* W = (g == 0) ? Wf : ((g == 1) ? Wi : Wh);
    wt[i] = (_Float16)W[k * Uc + n];              // wt[g][n][k]
}

// --- K1: gather + triple WMMA GEMM + gate epilogue + 16-step affine ---------
// Block = 16 consecutive timesteps of one batch row; 8 waves, wave w owns
// U columns [16w, 16w+16).
__global__ __launch_bounds__(256) void gates_scan_kernel(const int* __restrict__ sentence,
                                                         const _Float16* __restrict__ emb16,
                                                         const _Float16* __restrict__ wt16,
                                                         const float* __restrict__ b_f,
                                                         const float* __restrict__ b_i,
                                                         const float* __restrict__ b_h,
                                                         float* __restrict__ chF,
                                                         float* __restrict__ chI) {
    __shared__ _Float16 xs[16][Ec];               // 16 gathered f16 embedding rows (4 KB)

    const int tid     = threadIdx.x;
    const int rowbase = blockIdx.x * 16;          // = b*Tc + 16*chunk

    // Cooperative gather: 256 threads x 16B = 16 rows x 256B.
    {
        int r  = tid >> 4;
        int ch = tid & 15;
        int tok = sentence[rowbase + r];
        const uint4* src = (const uint4*)(emb16 + (size_t)tok * Ec) + ch;
        ((uint4*)(&xs[r][0]))[ch] = *src;
    }
    __syncthreads();

    const int lane = tid & 31;
    const int wave = tid >> 5;
    const int nloc = lane & 15;
    const int hsel = lane >> 4;                   // 0: t 0..7 of tile, 1: t 8..15
    const int col  = wave * 16 + nloc;            // global U column
    const int m0   = lane & 15;                   // A-matrix row held by this lane

    v8f cf  = {};
    v8f ci  = {};
    v8f chh = {};

    const _Float16* wf = wt16;
    const _Float16* wi = wt16 + Uc * Ec;
    const _Float16* wh = wt16 + 2 * Uc * Ec;

    #pragma unroll
    for (int kc = 0; kc < Ec / 32; ++kc) {
        // A fragment: lane<16 holds K {0..15},
        // lane>=16 holds K {16..31} of row (lane&15).
        AFrag a;
        const _Float16* arow = &xs[m0][kc * 32 + hsel * 16];
        a.h[0] = *(const v8h*)(arow);
        a.h[1] = *(const v8h*)(arow + 8);

        // B fragment: 16 contiguous K values of column `col` (32B, aligned).
        const size_t bo = (size_t)col * Ec + (size_t)kc * 32 + (size_t)hsel * 16;
        v16h bfrag_f = *(const v16h*)(wf + bo);
        v16h bfrag_i = *(const v16h*)(wi + bo);
        v16h bfrag_h = *(const v16h*)(wh + bo);

        cf  = __builtin_amdgcn_wmma_f32_16x16x32_f16(false, a.v, false, bfrag_f, (short)0, cf,  false, false);
        ci  = __builtin_amdgcn_wmma_f32_16x16x32_f16(false, a.v, false, bfrag_i, (short)0, ci,  false, false);
        chh = __builtin_amdgcn_wmma_f32_16x16x32_f16(false, a.v, false, bfrag_h, (short)0, chh, false, false);
    }

    // Gate epilogue + in-register 8-step affine compose.
    // C layout: VGPR r -> timestep (r + 8*hsel) within the tile, column col.
    const float bfv = b_f[col];
    const float biv = b_i[col];
    const float bhv = b_h[col];

    float F = 1.0f, I = 0.0f;
    #pragma unroll
    for (int r = 0; r < 8; ++r) {
        float fg  = 1.0f / (1.0f + __expf(-(cf[r]  + bfv)));
        float ig  = 1.0f / (1.0f + __expf(-(ci[r]  + biv)));
        float ht  = chh[r] + bhv;
        float d   = fg + ig;
        float fp  = fg / d;
        float inj = (ig / d) * ht;
        F = fp * F;                   // compose: h <- fp*h + inj, r ascending = t
        I = fp * I + inj;
    }

    // Merge low (t 0..7) and high (t 8..15) half-affines across lane pairs.
    float Fo = __shfl_xor(F, 16, 32);
    float Io = __shfl_xor(I, 16, 32);
    float Flo = hsel ? Fo : F;
    float Ilo = hsel ? Io : I;
    float Fhi = hsel ? F  : Fo;
    float Ihi = hsel ? I  : Io;
    float Fc = Fhi * Flo;
    float Ic = Fhi * Ilo + Ihi;

    if (lane < 16) {
        int b     = rowbase >> 11;                // / Tc
        int chunk = (rowbase & (Tc - 1)) >> 4;    // 16-step chunk index
        size_t o  = ((size_t)b * NCH + chunk) * Uc + col;
        chF[o] = Fc;
        chI[o] = Ic;
    }
}

// --- K2: combine 8 consecutive chunk affines --------------------------------
__global__ __launch_bounds__(256) void combine8_kernel(const float* __restrict__ chF,
                                                       const float* __restrict__ chI,
                                                       float* __restrict__ gF,
                                                       float* __restrict__ gI) {
    int tid = blockIdx.x * 256 + threadIdx.x;     // B*U*NGRP = 131072 threads
    int u = tid & (Uc - 1);
    int g = (tid >> 7) & (NGRP - 1);
    int b = tid >> 11;
    size_t base = ((size_t)b * NCH + (size_t)g * 8) * Uc + u;
    float F = 1.0f, I = 0.0f;
    #pragma unroll
    for (int k = 0; k < 8; ++k) {
        float f = chF[base];
        float j = chI[base];
        F = f * F;
        I = f * I + j;
        base += Uc;
    }
    int o = (b * NGRP + g) * Uc + u;
    gF[o] = F;
    gI[o] = I;
}

// --- K3: combine 16 group affines -> h_last ---------------------------------
__global__ __launch_bounds__(256) void combine_final_kernel(const float* __restrict__ gF,
                                                            const float* __restrict__ gI,
                                                            float* __restrict__ hlast) {
    int tid = blockIdx.x * 256 + threadIdx.x;     // B*U = 8192 threads
    int u = tid & (Uc - 1);
    int b = tid >> 7;
    float h = 0.0f;                               // h0 = 0
    #pragma unroll
    for (int c = 0; c < NGRP; ++c) {
        int o = (b * NGRP + c) * Uc + u;
        h = gF[o] * h + gI[o];
    }
    hlast[tid] = h;
}

// --- K4: classifier ---------------------------------------------------------
__global__ __launch_bounds__(256) void classifier_kernel(const float* __restrict__ hlast,
                                                         const float* __restrict__ Wc1,
                                                         const float* __restrict__ bc1,
                                                         const float* __restrict__ Wc2,
                                                         const float* __restrict__ bc2,
                                                         float* __restrict__ out) {
    __shared__ float z[Bc * 64];
    int tid = threadIdx.x;
    for (int e = tid; e < Bc * 64; e += 256) {
        int b = e >> 6;
        int j = e & 63;
        float s = bc1[j];
        #pragma unroll 4
        for (int u = 0; u < Uc; ++u) s += hlast[b * Uc + u] * Wc1[u * 64 + j];
        z[e] = s;
    }
    __syncthreads();
    if (tid < Bc) {
        float s = bc2[0];
        #pragma unroll 4
        for (int j = 0; j < 64; ++j) s += z[tid * 64 + j] * Wc2[j];
        out[tid] = 1.0f / (1.0f + __expf(-s));
    }
}

// ---------------------------------------------------------------------------
extern "C" void kernel_launch(void* const* d_in, const int* in_sizes, int n_in,
                              void* d_out, int out_size, void* d_ws, size_t ws_size,
                              hipStream_t stream) {
    const int*   sentence = (const int*)  d_in[0];
    const float* emb      = (const float*)d_in[1];
    const float* W_f      = (const float*)d_in[2];
    const float* b_f      = (const float*)d_in[3];
    const float* W_i      = (const float*)d_in[4];
    const float* b_i      = (const float*)d_in[5];
    const float* W_h      = (const float*)d_in[6];
    const float* b_h      = (const float*)d_in[7];
    const float* Wc1      = (const float*)d_in[8];
    const float* bc1      = (const float*)d_in[9];
    const float* Wc2      = (const float*)d_in[10];
    const float* bc2      = (const float*)d_in[11];
    float* out            = (float*)d_out;

    // Workspace layout (256B-aligned offsets), total ~22.4 MB.
    char* ws = (char*)d_ws;
    const size_t emb16_off = 0;
    const size_t emb16_sz  = (size_t)Vc * Ec * 2;                 // 12,865,792
    const size_t wt16_off  = emb16_off + emb16_sz;
    const size_t wt16_sz   = (size_t)3 * Uc * Ec * 2;             //     98,304
    const size_t chF_off   = wt16_off + wt16_sz;
    const size_t ch_sz     = (size_t)Bc * NCH * Uc * 4;           //  4,194,304
    const size_t chI_off   = chF_off + ch_sz;
    const size_t gF_off    = chI_off + ch_sz;
    const size_t g_sz      = (size_t)Bc * NGRP * Uc * 4;          //    524,288
    const size_t gI_off    = gF_off + g_sz;
    const size_t hlast_off = gI_off + g_sz;

    _Float16* emb16 = (_Float16*)(ws + emb16_off);
    _Float16* wt16  = (_Float16*)(ws + wt16_off);
    float*    chF   = (float*)(ws + chF_off);
    float*    chI   = (float*)(ws + chI_off);
    float*    gF    = (float*)(ws + gF_off);
    float*    gI    = (float*)(ws + gI_off);
    float*    hlast = (float*)(ws + hlast_off);

    const int nEmb = Vc * Ec;
    cvt_emb_kernel<<<(nEmb + 255) / 256, 256, 0, stream>>>(emb, emb16, nEmb);
    cvt_w_kernel<<<(3 * Uc * Ec) / 256, 256, 0, stream>>>(W_f, W_i, W_h, wt16);

    gates_scan_kernel<<<(Bc * Tc) / 16, 256, 0, stream>>>(sentence, emb16, wt16,
                                                          b_f, b_i, b_h, chF, chI);

    combine8_kernel<<<(Bc * Uc * NGRP) / 256, 256, 0, stream>>>(chF, chI, gF, gI);
    combine_final_kernel<<<(Bc * Uc) / 256, 256, 0, stream>>>(gF, gI, hlast);
    classifier_kernel<<<1, 256, 0, stream>>>(hlast, Wc1, bc1, Wc2, bc2, out);
}
